// SchNet_Classify_22196390986145
// MI455X (gfx1250) — matrix operand order repaired and checked by reading, not compiled, not run on hardware
//
#include <hip/hip_runtime.h>

#define N_ATOMS  262144
#define N_MOLS   8192
#define N_EDGES  4194304
#define HID      10
#define N_RBF    30
#define N_LAYERS 3
#define N_LABELS 11
#define CUTOFF   5.0f

typedef __attribute__((ext_vector_type(16))) _Float16 v16h;
typedef __attribute__((ext_vector_type(8)))  float    v8f;

// shifted softplus: softplus(x) - ln2 = ln2 * log2(0.5 + 0.5*2^(x*log2e)).
// Raw v_exp_f32 / v_log_f32 (base-2 HW ops) -> no denorm fixup, no branches:
// mul, exp, fma, log, mul, cndmask = 6 VALU total.
__device__ __forceinline__ float ssp(float x) {
    float t = __builtin_amdgcn_exp2f(x * 1.44269504088896f);
    float a = __builtin_amdgcn_logf(fmaf(0.5f, t, 0.5f)) * 0.69314718055995f;
    return (x > 15.0f) ? (x - 0.69314718056f) : a;
}

// ---------------------------------------------------------------- x = embedding[z]
__global__ void k_init_x(const int* __restrict__ z, const float* __restrict__ emb,
                         float* __restrict__ x) {
    int i = blockIdx.x * blockDim.x + threadIdx.x;
    if (i >= N_ATOMS * HID) return;
    int atom = i / HID, h = i - atom * HID;
    x[i] = emb[z[atom] * HID + h];
}

// ---------------------------------------------------------------- d, rcut per edge
__global__ void k_geom(const float* __restrict__ pos, const int* __restrict__ idx_i,
                       const int* __restrict__ idx_j, float* __restrict__ dist,
                       float* __restrict__ rc) {
    int e = blockIdx.x * blockDim.x + threadIdx.x;
    if (e >= N_EDGES) return;
    int i = idx_i[e], j = idx_j[e];
    float dx = pos[i * 3 + 0] - pos[j * 3 + 0];
    float dy = pos[i * 3 + 1] - pos[j * 3 + 1];
    float dz = pos[i * 3 + 2] - pos[j * 3 + 2];
    float d = sqrtf(dx * dx + dy * dy + dz * dz);
    dist[e] = d;
    float r = 0.5f * (__cosf(d * (3.14159265358979f / CUTOFF)) + 1.0f);
    rc[e] = (d < CUTOFF) ? r : 0.0f;
}

// ---------------------------------------------------------------- xf = x @ W ; agg = 0
__global__ void k_in2f(const float* __restrict__ x, const float* __restrict__ W,
                       float* __restrict__ xf, float* __restrict__ agg) {
    int atom = blockIdx.x * blockDim.x + threadIdx.x;
    if (atom >= N_ATOMS) return;
    float xi[HID];
#pragma unroll
    for (int k = 0; k < HID; ++k) xi[k] = x[atom * HID + k];
#pragma unroll
    for (int h = 0; h < HID; ++h) {
        float s = 0.0f;
#pragma unroll
        for (int k = 0; k < HID; ++k) s += xi[k] * W[k * HID + h];
        xf[atom * HID + h]  = s;
        agg[atom * HID + h] = 0.0f;
    }
}

// ---------------------------------------------------------------- fused edge kernel (WMMA)
// Per wave iteration: 32 edges as two independent 16-edge WMMA chains (ILP=2).
// Biases folded into the WMMAs via K zero-pad slots OUTSIDE the LDS-read range:
//   WMMA1: A[k=30]=1 (register const), B1[30][col]=b1[col] -> C1 = f_ij@W1 + b1
//   WMMA2: A[k=16]=1 (register const), B2[16][col]=b2[col] -> C2 = y@W2 + b2
// A2's LDS read stays one contiguous per-lane 16B block (k=0..7 / 8..15).
__global__ void __launch_bounds__(256) k_edge(
    const float* __restrict__ dist, const float* __restrict__ rc,
    const int* __restrict__ idx_i, const int* __restrict__ idx_j,
    const float* __restrict__ xf, float* __restrict__ agg,
    const float* __restrict__ W1, const float* __restrict__ b1v,
    const float* __restrict__ W2, const float* __restrict__ b2v) {
    const int lane = threadIdx.x & 31;
    const int wib  = threadIdx.x >> 5;
    const int col  = lane & 15;        // C/D and B column; also A row for this lane
    const bool hi  = lane >= 16;
    const int row  = col;              // A-matrix row owned by this lane

    __shared__ _Float16 xchg[8][2][16][20];  // per-wave, per-subtile transpose buffers

    // B fragments (loop-invariant): lane n<16 -> col n, K=0..7,16..23; lane n+16 -> col n, K=8..15,24..31
    // Bias rows live in K pad slots: K=30 for B1, K=16 for B2.
    v16h B1, B2;
#pragma unroll
    for (int t = 0; t < 16; ++t) {
        int k = t + ((t < 8) ? 0 : 8) + (hi ? 8 : 0);
        _Float16 w1 = (_Float16)0.0f, w2 = (_Float16)0.0f;
        if (col < HID) {
            w1 = (k < N_RBF) ? (_Float16)W1[k * HID + col]
                             : ((k == N_RBF) ? (_Float16)b1v[col] : (_Float16)0.0f);
            w2 = (k < HID)   ? (_Float16)W2[k * HID + col]
                             : ((k == 16)   ? (_Float16)b2v[col] : (_Float16)0.0f);
        }
        B1[t] = w1;
        B2[t] = w2;
    }

    const float width = CUTOFF / (float)(N_RBF - 1);
    const float coeff = -0.5f / (width * width);

    const int nTiles = N_EDGES / 32;
    const int wave   = blockIdx.x * (blockDim.x >> 5) + wib;
    const int nWaves = gridDim.x * (blockDim.x >> 5);

    for (int tile = wave; tile < nTiles; tile += nWaves) {
        const int base0 = tile * 32;
        const int base1 = base0 + 16;
        const float d0 = dist[base0 + row];
        const float d1 = dist[base1 + row];

        // A1: 16x32 f16, lane<16 holds K=0..7,16..23; lane>=16 holds K=8..15,24..31
        // K=30 slot carries 1.0 (bias row); K=31 is zero.
        v16h A1a, A1b;
#pragma unroll
        for (int t = 0; t < 16; ++t) {
            int k = t + ((t < 8) ? 0 : 8) + (hi ? 8 : 0);
            float mu = (float)k * width;
            float fa = d0 - mu, fb = d1 - mu;
            float ga = __builtin_amdgcn_exp2f(coeff * fa * fa * 1.44269504088896f);
            float gb = __builtin_amdgcn_exp2f(coeff * fb * fb * 1.44269504088896f);
            _Float16 pad = (k == N_RBF) ? (_Float16)1.0f : (_Float16)0.0f;
            A1a[t] = (k < N_RBF) ? (_Float16)ga : pad;
            A1b[t] = (k < N_RBF) ? (_Float16)gb : pad;
        }

        v8f C1a = {}, C1b = {};
        C1a = __builtin_amdgcn_wmma_f32_16x16x32_f16(false, A1a, false, B1,
                                                     (short)0, C1a, false, false);
        C1b = __builtin_amdgcn_wmma_f32_16x16x32_f16(false, A1b, false, B1,
                                                     (short)0, C1b, false, false);

        // ssp (bias already folded in), write C-layout -> LDS (row = edge, col = hid)
#pragma unroll
        for (int r = 0; r < 8; ++r) {
            int m = r + (hi ? 8 : 0);
            xchg[wib][0][m][col] = (_Float16)ssp(C1a[r]);
            xchg[wib][1][m][col] = (_Float16)ssp(C1b[r]);
        }

        // read back in A-layout: contiguous per-lane 16B block (k=0..7 or 8..15);
        // t=8 (K=16, low half only) = 1.0 bias row; rest zero.
        v16h A2a, A2b;
#pragma unroll
        for (int t = 0; t < 8; ++t) {
            int k = t + (hi ? 8 : 0);
            A2a[t] = xchg[wib][0][row][k];
            A2b[t] = xchg[wib][1][row][k];
        }
#pragma unroll
        for (int t = 8; t < 16; ++t) {
            _Float16 pad = (t == 8 && !hi) ? (_Float16)1.0f : (_Float16)0.0f;
            A2a[t] = pad;
            A2b[t] = pad;
        }

        v8f C2a = {}, C2b = {};
        C2a = __builtin_amdgcn_wmma_f32_16x16x32_f16(false, A2a, false, B2,
                                                     (short)0, C2a, false, false);
        C2b = __builtin_amdgcn_wmma_f32_16x16x32_f16(false, A2b, false, B2,
                                                     (short)0, C2b, false, false);

        // Wij = C2 * rcut ; msg = xf[j][col] * Wij ; agg[i][col] += msg
        if (col < HID) {
#pragma unroll
            for (int r = 0; r < 8; ++r) {
                int m = r + (hi ? 8 : 0);
                int ea = base0 + m;
                int eb = base1 + m;
                float wa = C2a[r] * rc[ea];
                float wb = C2b[r] * rc[eb];
                int ja = idx_j[ea], ia = idx_i[ea];
                int jb = idx_j[eb], ib = idx_i[eb];
                unsafeAtomicAdd(&agg[ia * HID + col], xf[ja * HID + col] * wa);
                unsafeAtomicAdd(&agg[ib * HID + col], xf[jb * HID + col] * wb);
            }
        }
    }
}

// ---------------------------------------------------------------- x += f2(agg)
__global__ void k_f2(const float* __restrict__ agg, const float* __restrict__ W1,
                     const float* __restrict__ b1, const float* __restrict__ W2,
                     const float* __restrict__ b2, float* __restrict__ x) {
    int atom = blockIdx.x * blockDim.x + threadIdx.x;
    if (atom >= N_ATOMS) return;
    float a[HID], t1[HID];
#pragma unroll
    for (int k = 0; k < HID; ++k) a[k] = agg[atom * HID + k];
#pragma unroll
    for (int h = 0; h < HID; ++h) {
        float s = b1[h];
#pragma unroll
        for (int k = 0; k < HID; ++k) s += a[k] * W1[k * HID + h];
        t1[h] = ssp(s);
    }
#pragma unroll
    for (int h = 0; h < HID; ++h) {
        float s = b2[h];
#pragma unroll
        for (int k = 0; k < HID; ++k) s += t1[k] * W2[k * HID + h];
        x[atom * HID + h] += s;
    }
}

// ---------------------------------------------------------------- readout
__global__ void k_zero(float* __restrict__ p, int n) {
    int i = blockIdx.x * blockDim.x + threadIdx.x;
    if (i < n) p[i] = 0.0f;
}

__global__ void k_scatter(const float* __restrict__ x, const int* __restrict__ batch,
                          float* __restrict__ hsum, float* __restrict__ cnt) {
    int atom = blockIdx.x * blockDim.x + threadIdx.x;
    if (atom >= N_ATOMS) return;
    int b = batch[atom];
    unsafeAtomicAdd(&cnt[b], 1.0f);
#pragma unroll
    for (int h = 0; h < HID; ++h)
        unsafeAtomicAdd(&hsum[b * HID + h], x[atom * HID + h]);
}

__global__ void k_cls(const float* __restrict__ hsum, const float* __restrict__ cnt,
                      const float* __restrict__ W, const float* __restrict__ b,
                      float* __restrict__ out) {
    int m = blockIdx.x * blockDim.x + threadIdx.x;
    if (m >= N_MOLS) return;
    float inv = 1.0f / fmaxf(cnt[m], 1.0f);
    float hr[HID];
#pragma unroll
    for (int k = 0; k < HID; ++k) hr[k] = hsum[m * HID + k] * inv;
#pragma unroll
    for (int c = 0; c < N_LABELS; ++c) {
        float s = b[c];
#pragma unroll
        for (int k = 0; k < HID; ++k) s += hr[k] * W[k * N_LABELS + c];
        out[m * N_LABELS + c] = s;
    }
}

// ================================================================ launch
extern "C" void kernel_launch(void* const* d_in, const int* in_sizes, int n_in,
                              void* d_out, int out_size, void* d_ws, size_t ws_size,
                              hipStream_t stream) {
    (void)in_sizes; (void)n_in; (void)out_size; (void)ws_size;
    const int*   z     = (const int*)d_in[0];
    const float* pos   = (const float*)d_in[1];
    const int*   batch = (const int*)d_in[2];
    const int*   idx_i = (const int*)d_in[3];
    const int*   idx_j = (const int*)d_in[4];
    const float* emb   = (const float*)d_in[5];
    const float* in2fW = (const float*)d_in[6];
    const float* fnW1  = (const float*)d_in[7];
    const float* fnb1  = (const float*)d_in[8];
    const float* fnW2  = (const float*)d_in[9];
    const float* fnb2  = (const float*)d_in[10];
    const float* f2W1  = (const float*)d_in[11];
    const float* f2b1  = (const float*)d_in[12];
    const float* f2W2  = (const float*)d_in[13];
    const float* f2b2  = (const float*)d_in[14];
    const float* clfW  = (const float*)d_in[15];
    const float* clfb  = (const float*)d_in[16];
    float* out = (float*)d_out;

    float* ws   = (float*)d_ws;
    float* dist = ws;                       // E
    float* rc   = dist + N_EDGES;           // E
    float* x    = rc   + N_EDGES;           // N*HID
    float* xf   = x    + N_ATOMS * HID;     // N*HID
    float* agg  = xf   + N_ATOMS * HID;     // N*HID
    float* hsum = agg  + N_ATOMS * HID;     // M*HID
    float* cnt  = hsum + N_MOLS * HID;      // M   (contiguous with hsum)

    k_init_x<<<(N_ATOMS * HID + 255) / 256, 256, 0, stream>>>(z, emb, x);
    k_geom<<<(N_EDGES + 255) / 256, 256, 0, stream>>>(pos, idx_i, idx_j, dist, rc);

    for (int l = 0; l < N_LAYERS; ++l) {
        k_in2f<<<(N_ATOMS + 255) / 256, 256, 0, stream>>>(x, in2fW + l * HID * HID, xf, agg);
        k_edge<<<2048, 256, 0, stream>>>(dist, rc, idx_i, idx_j, xf, agg,
                                         fnW1 + l * N_RBF * HID, fnb1 + l * HID,
                                         fnW2 + l * HID * HID,  fnb2 + l * HID);
        k_f2<<<(N_ATOMS + 255) / 256, 256, 0, stream>>>(agg, f2W1 + l * HID * HID,
                                                        f2b1 + l * HID,
                                                        f2W2 + l * HID * HID,
                                                        f2b2 + l * HID, x);
    }

    k_zero<<<(N_MOLS * (HID + 1) + 255) / 256, 256, 0, stream>>>(hsum, N_MOLS * (HID + 1));
    k_scatter<<<(N_ATOMS + 255) / 256, 256, 0, stream>>>(x, batch, hsum, cnt);
    k_cls<<<(N_MOLS + 255) / 256, 256, 0, stream>>>(hsum, cnt, clfW, clfb, out);
}
